// JointSpatioTemporalAttention_61667140436638
// MI455X (gfx1250) — compile-verified, matrix operand order checked by reading, not run on hardware
//
#include <hip/hip_runtime.h>

#define DEVINL __device__ __forceinline__

typedef __attribute__((ext_vector_type(16))) __bf16 v16bf;
typedef __attribute__((ext_vector_type(8)))  float  v8f;

union FragBF { v16bf v; uint4 u[2]; };

// ---- problem constants (fixed by the reference) ----
constexpr int kB   = 2;
constexpr int kT   = 8, kHs = 16, kWs = 16;
constexpr int kN   = kT * kHs * kWs;   // 2048
constexpr int kD   = 1024;
constexpr int kNH  = 16;
constexpr int kNKV = 4;
constexpr int kHD  = 64;
constexpr int kM   = kB * kN;          // 4096 rows
constexpr int kGQ  = kNH / kNKV;       // 4
constexpr float kScale = 0.125f;       // HD^-0.5
constexpr float kLogTheta = 9.210340371976184f; // ln(10000)

DEVINL unsigned short f2bf(float f) {
  union { float f; unsigned int u; } c; c.f = f;
  unsigned int r = 0x7FFFu + ((c.u >> 16) & 1u);
  return (unsigned short)((c.u + r) >> 16);
}

// Load a 16x32 bf16 A-style fragment. `base` points at the start of this
// lane's row within the current 32-wide K window (16B aligned).
DEVINL void load_frag(FragBF& f, const unsigned short* base, int hi) {
  f.u[0] = *reinterpret_cast<const uint4*>(base + hi * 8);
  f.u[1] = *reinterpret_cast<const uint4*>(base + 16 + hi * 8);
}

DEVINL v8f wmma_bf16(const FragBF& a, const FragBF& b, v8f c) {
  return __builtin_amdgcn_wmma_f32_16x16x32_bf16(false, a.v, false, b.v,
                                                 (short)0, c, false, false);
}

// Async copy 16B global -> LDS, tracked by ASYNCcnt.
DEVINL void async_copy_b128(unsigned lds_byte_off, const void* gptr) {
  asm volatile("global_load_async_to_lds_b128 %0, %1, off"
               :: "v"(lds_byte_off),
                  "v"((unsigned long long)(uintptr_t)gptr)
               : "memory");
}
DEVINL void wait_async_le4() {
  asm volatile("s_wait_asynccnt 0x4" ::: "memory");
}

// ---------------- elementwise conversions ----------------
__global__ void convert_f32_bf16(const float* __restrict__ in,
                                 unsigned short* __restrict__ out, int n) {
  int i = blockIdx.x * blockDim.x + threadIdx.x;
  if (i < n) out[i] = f2bf(in[i]);
}

// W is [K][Ncol] row-major fp32 -> Wt is [Ncol][K] bf16
__global__ void transpose_w(const float* __restrict__ W,
                            unsigned short* __restrict__ Wt, int K, int Ncol) {
  int i = blockIdx.x * blockDim.x + threadIdx.x;
  if (i >= K * Ncol) return;
  int n = i % Ncol;
  int k = i / Ncol;
  Wt[(size_t)n * K + k] = f2bf(W[i]);
}

// v fp32 [b][n][kv][64] -> vt bf16 [b][kv][64][n]
__global__ void prep_v(const float* __restrict__ v,
                       unsigned short* __restrict__ vt) {
  int i = blockIdx.x * blockDim.x + threadIdx.x;
  if (i >= kB * kN * kNKV * kHD) return;
  int d  = i & 63;
  int kv = (i >> 6) & 3;
  int n  = (i >> 8) & (kN - 1);
  int b  = i >> 19;
  vt[((size_t)(b * kNKV + kv) * kHD + d) * kN + n] = f2bf(v[i]);
}

// ---------------- bf16 WMMA GEMM: C = A * Wt^T + bias ----------------
// A: [M][K] bf16 row-major, Wt: [Ncol][K] bf16, C: [M][Ncol] fp32.
// grid = (Ncol/256, M/16), block = 128. Each wave computes a 16x64 strip
// (one A fragment feeds 4 WMMAs). The K-loop is software-pipelined with two
// explicit fragment sets so the next half-step's 10 loads are in flight
// under the current 4 WMMAs (partial loadcnt waits instead of 0x0).
__global__ void __launch_bounds__(128)
gemm_bf16(const unsigned short* __restrict__ A,
          const unsigned short* __restrict__ Wt,
          const float* __restrict__ bias,
          float* __restrict__ C, int Ncol, int K) {
  int lane = threadIdx.x & 31;
  int wid  = threadIdx.x >> 5;
  int col  = lane & 15;
  int hi   = lane >> 4;
  int row0 = blockIdx.y * 16;
  int colBase = blockIdx.x * 256 + wid * 64;

  const unsigned short* arow  = A  + (size_t)(row0 + col) * K;
  const unsigned short* brow0 = Wt + (size_t)(colBase + 0  + col) * K;
  const unsigned short* brow1 = Wt + (size_t)(colBase + 16 + col) * K;
  const unsigned short* brow2 = Wt + (size_t)(colBase + 32 + col) * K;
  const unsigned short* brow3 = Wt + (size_t)(colBase + 48 + col) * K;

  v8f acc0 = {}, acc1 = {}, acc2 = {}, acc3 = {};

  FragBF aA, b0A, b1A, b2A, b3A;   // set A (even half-steps)
  FragBF aB, b0B, b1B, b2B, b3B;   // set B (odd half-steps)

  // prologue: fill set A with k0 = 0
  load_frag(aA,  arow,  hi);
  load_frag(b0A, brow0, hi);
  load_frag(b1A, brow1, hi);
  load_frag(b2A, brow2, hi);
  load_frag(b3A, brow3, hi);

  for (int k0 = 0; k0 < K; k0 += 64) {   // K is a multiple of 64 here
    // issue set B loads (k0+32) before consuming set A
    load_frag(aB,  arow  + k0 + 32, hi);
    load_frag(b0B, brow0 + k0 + 32, hi);
    load_frag(b1B, brow1 + k0 + 32, hi);
    load_frag(b2B, brow2 + k0 + 32, hi);
    load_frag(b3B, brow3 + k0 + 32, hi);
    if (k0 + 256 < K) {
      __builtin_prefetch(arow + k0 + 256, 0, 3);
      __builtin_prefetch(brow0 + k0 + 256, 0, 3);
    }
    acc0 = wmma_bf16(aA, b0A, acc0);
    acc1 = wmma_bf16(aA, b1A, acc1);
    acc2 = wmma_bf16(aA, b2A, acc2);
    acc3 = wmma_bf16(aA, b3A, acc3);

    // issue set A loads (k0+64) before consuming set B
    if (k0 + 64 < K) {
      load_frag(aA,  arow  + k0 + 64, hi);
      load_frag(b0A, brow0 + k0 + 64, hi);
      load_frag(b1A, brow1 + k0 + 64, hi);
      load_frag(b2A, brow2 + k0 + 64, hi);
      load_frag(b3A, brow3 + k0 + 64, hi);
    }
    acc0 = wmma_bf16(aB, b0B, acc0);
    acc1 = wmma_bf16(aB, b1B, acc1);
    acc2 = wmma_bf16(aB, b2B, acc2);
    acc3 = wmma_bf16(aB, b3B, acc3);
  }

  v8f* accs[4] = { &acc0, &acc1, &acc2, &acc3 };
#pragma unroll
  for (int t = 0; t < 4; t++) {
    int colt = colBase + t * 16 + col;
    float bb = bias[colt];
#pragma unroll
    for (int r = 0; r < 8; r++) {
      int row = row0 + r + hi * 8;
      C[(size_t)row * Ncol + colt] = (*accs[t])[r] + bb;
    }
  }
}

// ---------------- RMSNorm + factored 3D RoPE ----------------
DEVINL float rope_one(const float* xs, int d, int t, int hh, int w) {
  int local, dd, pos, base;
  if (d < 16)      { local = d;      dd = 16; pos = t;  base = 0;  }
  else if (d < 32) { local = d - 16; dd = 16; pos = hh; base = 16; }
  else             { local = d - 32; dd = 32; pos = w;  base = 32; }
  int half = dd >> 1;
  int i = local & (half - 1);
  float freq = __expf(-(2.0f * (float)i / (float)dd) * kLogTheta);
  float a = (float)pos * freq;
  float cs = __cosf(a);
  float sn = __sinf(a);
  float val = xs[d];
  float other = xs[base + ((local < half) ? (local + half) : (local - half))];
  return (local < half) ? (val * cs - other * sn) : (val * cs + other * sn);
}

__global__ void __launch_bounds__(32)
prep_qk(const float* __restrict__ src, const float* __restrict__ nw,
        unsigned short* __restrict__ dst, int nh) {
  int idx  = blockIdx.x;            // B*N*nh
  int head = idx % nh;
  int row  = idx / nh;              // 0..B*N-1
  int b    = row / kN;
  int n    = row % kN;
  int lane = threadIdx.x;

  const float* p = src + (size_t)row * (nh * kHD) + head * kHD;
  float x0 = p[lane];
  float x1 = p[lane + 32];
  float ss = x0 * x0 + x1 * x1;
#pragma unroll
  for (int m = 16; m; m >>= 1) ss += __shfl_xor(ss, m, 32);
  float rinv = rsqrtf(ss * (1.0f / 64.0f) + 1e-6f);

  __shared__ float xs[64];
  xs[lane]      = x0 * rinv * nw[lane];
  xs[lane + 32] = x1 * rinv * nw[lane + 32];
  __syncthreads();

  int t  = n >> 8;
  int hh = (n >> 4) & 15;
  int w  = n & 15;
  float o0 = rope_one(xs, lane,      t, hh, w);
  float o1 = rope_one(xs, lane + 32, t, hh, w);

  size_t obase = ((size_t)(b * nh + head) * kN + n) * kHD;
  dst[obase + lane]      = f2bf(o0);
  dst[obase + lane + 32] = f2bf(o1);
}

// ---------------- fused flash attention ----------------
// qb bf16 [b][h][n][64], kb bf16 [b][kv][n][64], vt bf16 [b][kv][64][n]
// attnb bf16 [B*N][D]. 4 waves/block = 4 consecutive q-tiles of the SAME
// (b, head) -> K/V tiles are staged once per block in LDS via double-buffered
// GLOBAL_LOAD_ASYNC_TO_LDS_B128 (ASYNCcnt), overlapping DMA with WMMA+softmax.
__global__ void __launch_bounds__(128)
attn_kernel(const unsigned short* __restrict__ qb,
            const unsigned short* __restrict__ kb,
            const unsigned short* __restrict__ vt,
            unsigned short* __restrict__ attnb) {
  int tid  = threadIdx.x;
  int lane = tid & 31;
  int wid  = tid >> 5;
  int gt   = blockIdx.x * 4 + wid;          // global q-tile id
  const int tilesPerHead = kN / 16;         // 128 (multiple of 4)
  int b   = gt / (kNH * tilesPerHead);
  int rem = gt % (kNH * tilesPerHead);
  int h   = rem / tilesPerHead;
  int qt  = rem % tilesPerHead;
  int kv  = h / kGQ;
  int col = lane & 15;
  int hi  = lane >> 4;

  // bytes: buf0 {K 4KB, V 4KB} @0, buf1 @8192, pbuf 4KB @16384
  __shared__ alignas(128) unsigned short smem[2 * 4096 + 2048];
  unsigned smemBase = (unsigned)(uintptr_t)(const void*)smem;
  unsigned short* pbuf = smem + 8192 + wid * 512;

  // Q fragments (loaded once)
  const unsigned short* qrow =
      qb + ((size_t)(b * kNH + h) * kN + qt * 16 + col) * kHD;
  FragBF qf0, qf1;
  load_frag(qf0, qrow + 0,  hi);
  load_frag(qf1, qrow + 32, hi);

  const unsigned short* kbase = kb + (size_t)(b * kNKV + kv) * kN * kHD;
  const unsigned short* vbase = vt + (size_t)(b * kNKV + kv) * kHD * kN;

  // per-thread staging slots: each thread moves 4x16B per tile pair.
  // K tile [key][64]: chunkA keys 0..15, chunkB keys 16..31; lds off = tid*16.
  // V tile [d][32]:   chunkA d 0..31,    chunkB d 32..63;    lds off = tid*16.
  const unsigned short* kSrcA = kbase + (size_t)(tid >> 3) * kHD + (tid & 7) * 8;
  const unsigned short* kSrcB = kSrcA + (size_t)16 * kHD;
  const unsigned short* vSrcA = vbase + (size_t)(tid >> 2) * kN + (tid & 3) * 8;
  const unsigned short* vSrcB = vSrcA + (size_t)32 * kN;

  auto issue = [&](int buf, int jn) {
    unsigned base = smemBase + (unsigned)buf * 8192u + (unsigned)tid * 16u;
    async_copy_b128(base,        kSrcA + (size_t)jn * kHD);
    async_copy_b128(base + 2048, kSrcB + (size_t)jn * kHD);
    async_copy_b128(base + 4096, vSrcA + jn);
    async_copy_b128(base + 6144, vSrcB + jn);
  };

  v8f o0 = {}, o1 = {}, o2 = {}, o3 = {};
  float mrow[8], lrow[8];
#pragma unroll
  for (int r = 0; r < 8; r++) { mrow[r] = -__builtin_inff(); lrow[r] = 0.0f; }

  issue(0, 0);  // prologue: buf0 <- keys [0,32)

  const int iters = kN / 32;  // 64
  for (int i = 0; i < iters; i++) {
    int buf = i & 1;
    // prefetch next tile pair into the other buffer (wraps harmlessly at end)
    issue(buf ^ 1, ((i + 1) & (iters - 1)) * 32);
    wait_async_le4();    // retire current buffer's 4 ops, next 4 keep flying
    __syncthreads();

    const unsigned short* kT = smem + buf * 4096;
    const unsigned short* vT = smem + buf * 4096 + 2048;

    // ---- scores for 32 keys: two 16x16 tiles, K=64 via 2 WMMAs each ----
    v8f s0 = {}, s1 = {};
    {
      const unsigned short* k0r = kT + col * kHD;
      const unsigned short* k1r = kT + (16 + col) * kHD;
      FragBF kf0, kf1, kf2, kf3;
      load_frag(kf0, k0r + 0,  hi);
      load_frag(kf1, k0r + 32, hi);
      load_frag(kf2, k1r + 0,  hi);
      load_frag(kf3, k1r + 32, hi);
      s0 = wmma_bf16(qf0, kf0, s0);
      s0 = wmma_bf16(qf1, kf1, s0);
      s1 = wmma_bf16(qf0, kf2, s1);
      s1 = wmma_bf16(qf1, kf3, s1);
    }

    // ---- online softmax; row r of this half = r + hi*8 ----
    float alpha[8];
#pragma unroll
    for (int r = 0; r < 8; r++) {
      float a  = s0[r] * kScale;
      float bb = s1[r] * kScale;
      float mx = fmaxf(a, bb);
#pragma unroll
      for (int m = 8; m; m >>= 1) mx = fmaxf(mx, __shfl_xor(mx, m, 32));
      float mn = fmaxf(mrow[r], mx);
      float al = __expf(mrow[r] - mn);
      float p0 = __expf(a - mn);
      float p1 = __expf(bb - mn);
      float rs = p0 + p1;
#pragma unroll
      for (int m = 8; m; m >>= 1) rs += __shfl_xor(rs, m, 32);
      lrow[r] = lrow[r] * al + rs;
      mrow[r] = mn;
      alpha[r] = al;
      int row = r + hi * 8;
      pbuf[row * 32 + col]      = f2bf(p0);
      pbuf[row * 32 + col + 16] = f2bf(p1);
    }
#pragma unroll
    for (int r = 0; r < 8; r++) {
      o0[r] *= alpha[r]; o1[r] *= alpha[r];
      o2[r] *= alpha[r]; o3[r] *= alpha[r];
    }

    __syncthreads();
    // reshape P (acc layout) -> A-fragment layout via LDS
    FragBF pf;
    const unsigned short* prow = pbuf + (size_t)col * 32;
    pf.u[0] = *reinterpret_cast<const uint4*>(prow + hi * 8);
    pf.u[1] = *reinterpret_cast<const uint4*>(prow + 16 + hi * 8);

    // ---- O += P * V, 4 output d-tiles of 16; distinct V fragments ----
    FragBF vf0, vf1, vf2, vf3;
    load_frag(vf0, vT + (0  + col) * 32, hi);
    load_frag(vf1, vT + (16 + col) * 32, hi);
    load_frag(vf2, vT + (32 + col) * 32, hi);
    load_frag(vf3, vT + (48 + col) * 32, hi);
    o0 = wmma_bf16(pf, vf0, o0);
    o1 = wmma_bf16(pf, vf1, o1);
    o2 = wmma_bf16(pf, vf2, o2);
    o3 = wmma_bf16(pf, vf3, o3);

    __syncthreads();  // all reads of buf done before it is refilled
  }

  // ---- epilogue: divide by l, write bf16 [B*N][D] ----
#pragma unroll
  for (int r = 0; r < 8; r++) {
    float inv = 1.0f / lrow[r];
    int n = qt * 16 + r + hi * 8;
    size_t ob = ((size_t)(b * kN + n)) * kD + h * kHD;
    attnb[ob + 0  + col] = f2bf(o0[r] * inv);
    attnb[ob + 16 + col] = f2bf(o1[r] * inv);
    attnb[ob + 32 + col] = f2bf(o2[r] * inv);
    attnb[ob + 48 + col] = f2bf(o3[r] * inv);
  }
}

// ---------------- host ----------------
extern "C" void kernel_launch(void* const* d_in, const int* in_sizes, int n_in,
                              void* d_out, int out_size, void* d_ws, size_t ws_size,
                              hipStream_t stream) {
  (void)in_sizes; (void)n_in; (void)out_size; (void)ws_size;
  const float* x  = (const float*)d_in[0];
  const float* Wq = (const float*)d_in[1];
  const float* bq = (const float*)d_in[2];
  const float* Wk = (const float*)d_in[3];
  const float* bk = (const float*)d_in[4];
  const float* Wv = (const float*)d_in[5];
  const float* bv = (const float*)d_in[6];
  const float* Wo = (const float*)d_in[7];
  const float* bo = (const float*)d_in[8];
  const float* qn = (const float*)d_in[9];
  const float* kn = (const float*)d_in[10];
  float* out = (float*)d_out;

  char* ws = (char*)d_ws;
  size_t off = 0;
  auto walloc = [&](size_t bytes) {
    char* p = ws + off;
    off += (bytes + 255) & ~(size_t)255;
    return p;
  };

  unsigned short* xb    = (unsigned short*)walloc((size_t)kM * kD * 2);
  unsigned short* wqT   = (unsigned short*)walloc((size_t)kD * kD * 2);
  unsigned short* wkT   = (unsigned short*)walloc((size_t)(kNKV * kHD) * kD * 2);
  unsigned short* wvT   = (unsigned short*)walloc((size_t)(kNKV * kHD) * kD * 2);
  unsigned short* woT   = (unsigned short*)walloc((size_t)kD * kD * 2);
  float*          qf    = (float*)walloc((size_t)kM * kD * 4);
  float*          kf    = (float*)walloc((size_t)kM * (kNKV * kHD) * 4);
  float*          vf    = (float*)walloc((size_t)kM * (kNKV * kHD) * 4);
  unsigned short* qbb   = (unsigned short*)walloc((size_t)kB * kNH * kN * kHD * 2);
  unsigned short* kbb   = (unsigned short*)walloc((size_t)kB * kNKV * kN * kHD * 2);
  unsigned short* vtt   = (unsigned short*)walloc((size_t)kB * kNKV * kHD * kN * 2);
  unsigned short* attnb = (unsigned short*)walloc((size_t)kM * kD * 2);

  const int kvD = kNKV * kHD;  // 256
  int tot;

  tot = kM * kD;
  convert_f32_bf16<<<(tot + 255) / 256, 256, 0, stream>>>(x, xb, tot);
  tot = kD * kD;
  transpose_w<<<(tot + 255) / 256, 256, 0, stream>>>(Wq, wqT, kD, kD);
  tot = kD * kvD;
  transpose_w<<<(tot + 255) / 256, 256, 0, stream>>>(Wk, wkT, kD, kvD);
  transpose_w<<<(tot + 255) / 256, 256, 0, stream>>>(Wv, wvT, kD, kvD);
  tot = kD * kD;
  transpose_w<<<(tot + 255) / 256, 256, 0, stream>>>(Wo, woT, kD, kD);

  // QKV projections
  gemm_bf16<<<dim3(kD / 256,  kM / 16), 128, 0, stream>>>(xb, wqT, bq, qf, kD,  kD);
  gemm_bf16<<<dim3(kvD / 256, kM / 16), 128, 0, stream>>>(xb, wkT, bk, kf, kvD, kD);
  gemm_bf16<<<dim3(kvD / 256, kM / 16), 128, 0, stream>>>(xb, wvT, bv, vf, kvD, kD);

  // norm + rope + layout
  prep_qk<<<kM * kNH,  32, 0, stream>>>(qf, qn, qbb, kNH);
  prep_qk<<<kM * kNKV, 32, 0, stream>>>(kf, kn, kbb, kNKV);
  tot = kB * kN * kNKV * kHD;
  prep_v<<<(tot + 255) / 256, 256, 0, stream>>>(vf, vtt);

  // fused attention: one 16-row q tile per wave, LDS-staged K/V per block
  attn_kernel<<<(kB * kNH * (kN / 16)) / 4, 128, 0, stream>>>(qbb, kbb, vtt, attnb);

  // output projection
  gemm_bf16<<<dim3(kD / 256, kM / 16), 128, 0, stream>>>(attnb, woT, bo, out, kD, kD);
}
